// MultiHeadAttention_22256520528656
// MI455X (gfx1250) — compile-verified, matrix operand order checked by reading, not run on hardware
//
#include <hip/hip_runtime.h>
#include <hip/hip_bf16.h>

// ---------------------------------------------------------------------------
// MI455X (gfx1250) multi-head attention with sliding-window band mask.
// - All matmuls via v_wmma_f32_16x16x32_bf16 (fp32 accumulate)
// - GEMM tile staging via Tensor Data Mover (tensor_load_to_lds + TENSORcnt)
// - Operands pre-converted to bf16 once (weights transposed to [N][K])
// ---------------------------------------------------------------------------

typedef __attribute__((ext_vector_type(16))) __bf16 v16bf;
typedef __attribute__((ext_vector_type(8)))  __bf16 v8bf;
typedef __attribute__((ext_vector_type(4)))  __bf16 v4bf;
typedef __attribute__((ext_vector_type(8)))  float  v8f;
typedef __attribute__((ext_vector_type(4)))  unsigned int v4u;
typedef __attribute__((ext_vector_type(8)))  int v8i;
typedef __attribute__((ext_vector_type(4)))  int v4i;

#define S_LEN 2048
#define DMODEL 1024
#define NH 16
#define DK 64
#define WIN 256
#define SMAX 544   // 34 key tiles of 16 (window is <= 528 keys, pad to 32)

__device__ __forceinline__ v8f zero_v8f() { v8f z = {}; return z; }

__device__ __forceinline__ v16bf combine16(v8bf lo, v8bf hi) {
    v16bf r;
#pragma unroll
    for (int j = 0; j < 8; ++j) { r[j] = lo[j]; r[8 + j] = hi[j]; }
    return r;
}

__device__ __forceinline__ v8f wmma_bf16(v16bf a, v16bf b, v8f c) {
    // 8 args: (neg_a, A, neg_b, B, c_mod, C, reuse_a, reuse_b)
    return __builtin_amdgcn_wmma_f32_16x16x32_bf16(false, a, false, b,
                                                   (short)0, c, false, false);
}

// ---------------------------------------------------------------------------
// TDM: issue a 2D bf16 tile load (rows x cols, row stride = stride0 elements)
// into LDS at byte offset lds_off. Descriptor packing per CDNA5 ISA 8.3/8.4:
//   group0: count=1 | lds_addr | global_addr(57b) | type=2
//   group1: data_size=2B, tensor_dim0/1, tile_dim0/1, tensor_dim0_stride
// 6-arg builtin form (clang-23 / therock headers): (g0, g1, g2, g3, g4, cpol)
// ---------------------------------------------------------------------------
__device__ __forceinline__ void tdm_load_2d(unsigned lds_off, const void* gaddr,
                                            unsigned tensor_d0, unsigned tensor_d1,
                                            unsigned tile_d0, unsigned tile_d1,
                                            unsigned stride0)
{
    unsigned long long ga = (unsigned long long)(uintptr_t)gaddr;
    v4u g0;
    g0.x = 1u;                                            // count=1 (valid)
    g0.y = lds_off;                                       // lds_addr [63:32]
    g0.z = (unsigned)(ga & 0xFFFFFFFFu);                  // global_addr lo
    g0.w = (unsigned)((ga >> 32) & 0x1FFFFFFu) | (2u << 30);  // addr hi | type=2
    v8i g1;
    g1[0] = (int)(1u << 16);                              // data_size=1 -> 2 bytes
    g1[1] = (int)((tensor_d0 & 0xFFFFu) << 16);           // tensor_dim0[15:0]
    g1[2] = (int)((tensor_d0 >> 16) | ((tensor_d1 & 0xFFFFu) << 16));
    g1[3] = (int)((tensor_d1 >> 16) | (tile_d0 << 16));   // tile_dim0
    g1[4] = (int)tile_d1;                                 // tile_dim1 (tile_dim2=0)
    g1[5] = (int)stride0;                                 // tensor_dim0_stride lo
    g1[6] = 0;
    g1[7] = 0;
    v4i z4 = {};
    v8i z8 = {};
    __builtin_amdgcn_tensor_load_to_lds(g0, g1, z4, z4, z8, 0);
}

// ---------------------------------------------------------------------------
// fp32 -> bf16 bulk convert (4 elements / thread)
// ---------------------------------------------------------------------------
__global__ __launch_bounds__(256) void cvt_bf16_kernel(
    const float* __restrict__ src, __bf16* __restrict__ dst, int n)
{
    int i = (blockIdx.x * 256 + threadIdx.x) * 4;
    if (i + 3 < n) {
        float4 f = *(const float4*)(src + i);
        v4bf o;
        o[0] = (__bf16)f.x; o[1] = (__bf16)f.y;
        o[2] = (__bf16)f.z; o[3] = (__bf16)f.w;
        *(v4bf*)(dst + i) = o;
    }
}

// ---------------------------------------------------------------------------
// W[K,N] fp32 -> Wt[N,K] bf16 (32x32 LDS transpose tiles)
// ---------------------------------------------------------------------------
__global__ __launch_bounds__(256) void wt_kernel(
    const float* __restrict__ W, __bf16* __restrict__ Wt, int N, int K)
{
    __shared__ __bf16 t[32][33];
    const int n0 = blockIdx.x * 32, k0 = blockIdx.y * 32;
    const int tx = threadIdx.x & 31, ty = threadIdx.x >> 5;   // 32 x 8
#pragma unroll
    for (int r = 0; r < 32; r += 8)
        t[ty + r][tx] = (__bf16)W[(size_t)(k0 + ty + r) * N + n0 + tx];
    __syncthreads();
#pragma unroll
    for (int r = 0; r < 32; r += 8)
        Wt[(size_t)(n0 + ty + r) * K + k0 + tx] = t[tx][ty + r];
}

// ---------------------------------------------------------------------------
// Tiled GEMM with bias: C[M,N] = A[M,K] @ Wt[N,K]^T + bias[N]
// A, Wt in bf16. 128x128 block tile, BK=32, 256 threads (8 waves),
// wave tile 32x64. Both LDS tiles are staged by the Tensor Data Mover.
// ---------------------------------------------------------------------------
template <typename TOut>
__global__ __launch_bounds__(256) void gemm_tdm_kernel(
    const __bf16* __restrict__ A, const __bf16* __restrict__ Wt,
    const float* __restrict__ bias, TOut* __restrict__ C,
    int M, int N, int K)
{
    __shared__ __bf16 As[128 * 32];   // [m][k], 8 KB
    __shared__ __bf16 Bs[128 * 32];   // [n][k], 8 KB

    const int tid  = threadIdx.x;
    const int lane = tid & 31;
    const int wave = tid >> 5;
    const int half = lane >> 4;
    const int sub  = lane & 15;

    const int m0 = blockIdx.y * 128;
    const int n0 = blockIdx.x * 128;

    const int rbase = (wave >> 1) * 32;  // 4 waves along M
    const int cbase = (wave & 1) * 64;   // 2 waves along N

    const unsigned lds_a = (unsigned)(uintptr_t)(void*)&As[0];
    const unsigned lds_b = (unsigned)(uintptr_t)(void*)&Bs[0];

    v8f acc[2][4];
#pragma unroll
    for (int i = 0; i < 2; ++i)
#pragma unroll
        for (int j = 0; j < 4; ++j) acc[i][j] = zero_v8f();

    for (int k0 = 0; k0 < K; k0 += 32) {
        __syncthreads();   // previous iteration's reads done before overwrite
        if (wave == 0) {   // TDM ignores EXEC; issue from a single wave
            tdm_load_2d(lds_a, A + (size_t)m0 * K + k0,
                        (unsigned)K, (unsigned)(M - m0), 32u, 128u, (unsigned)K);
            tdm_load_2d(lds_b, Wt + (size_t)n0 * K + k0,
                        (unsigned)K, (unsigned)(N - n0), 32u, 128u, (unsigned)K);
            __builtin_amdgcn_s_wait_tensorcnt(0);
        }
        __syncthreads();   // LDS tiles visible to all waves

        v16bf afrag[2], bfrag[4];
#pragma unroll
        for (int i = 0; i < 2; ++i) {
            int r = rbase + i * 16 + sub;
            v8bf lo = *(const v8bf*)&As[r * 32 + half * 8];
            v8bf hi = *(const v8bf*)&As[r * 32 + 16 + half * 8];
            afrag[i] = combine16(lo, hi);
        }
#pragma unroll
        for (int j = 0; j < 4; ++j) {
            int nn = cbase + j * 16 + sub;
            v8bf lo = *(const v8bf*)&Bs[nn * 32 + half * 8];
            v8bf hi = *(const v8bf*)&Bs[nn * 32 + 16 + half * 8];
            bfrag[j] = combine16(lo, hi);
        }
#pragma unroll
        for (int i = 0; i < 2; ++i)
#pragma unroll
            for (int j = 0; j < 4; ++j)
                acc[i][j] = wmma_bf16(afrag[i], bfrag[j], acc[i][j]);
    }

    // Epilogue: C/D layout -> lane holds column (n), 8 rows per VGPR set.
#pragma unroll
    for (int i = 0; i < 2; ++i) {
#pragma unroll
        for (int j = 0; j < 4; ++j) {
            int n = n0 + cbase + j * 16 + sub;
            float bv = bias[n];
#pragma unroll
            for (int r = 0; r < 8; ++r) {
                int m = m0 + rbase + i * 16 + half * 8 + r;
                C[(size_t)m * N + n] = (TOut)(acc[i][j][r] + bv);
            }
        }
    }
}

// ---------------------------------------------------------------------------
// Banded attention. One block = (batch b, head h, 16-query tile).
// Qh/Kh/Vh/ctx layout: [B, S, H, DK] (d contiguous).
// ---------------------------------------------------------------------------
__global__ __launch_bounds__(128) void attn_kernel(
    const __bf16* __restrict__ Qh, const __bf16* __restrict__ Kh,
    const __bf16* __restrict__ Vh, float* __restrict__ attn,
    __bf16* __restrict__ ctx)
{
    __shared__ float  scores[16 * SMAX];  // 34816 B
    __shared__ __bf16 pbuf[16 * SMAX];    // 17408 B
    __shared__ __bf16 vts[64 * 32];       // [d][key-in-chunk], 4096 B

    const int tid  = threadIdx.x;
    const int lane = tid & 31;
    const int wave = tid >> 5;
    const int half = lane >> 4;
    const int sub  = lane & 15;

    const int qt = blockIdx.x & 127;          // S/16 = 128 query tiles
    const int h  = (blockIdx.x >> 7) & 15;
    const int b  = blockIdx.x >> 11;

    const int q0     = qt * 16;
    const int kstart = max(0, q0 - WIN);               // multiple of 16
    const int kend   = min(S_LEN, q0 + 16 + WIN);      // multiple of 16
    const int nk     = kend - kstart;                  // <= 528
    const int nkt    = nk >> 4;
    const int nkp    = (nk + 31) & ~31;                // <= 544

    for (int i = tid; i < 16 * SMAX; i += 128) pbuf[i] = (__bf16)0.0f;

    // Q fragments: 16x64 split into two 16x32 bf16 A-fragments.
    v16bf qf[2];
    {
        const size_t qbase = ((size_t)(b * S_LEN + q0 + sub) * NH + h) * DK;
#pragma unroll
        for (int c2 = 0; c2 < 2; ++c2) {
            v8bf lo = *(const v8bf*)&Qh[qbase + c2 * 32 + half * 8];
            v8bf hi = *(const v8bf*)&Qh[qbase + c2 * 32 + 16 + half * 8];
            qf[c2] = combine16(lo, hi);
        }
    }

    // Phase 1: scores = (Q K^T) / sqrt(dk) over the key window.
    for (int kt = wave; kt < nkt; kt += 4) {
        const int kb = kstart + kt * 16;
        const size_t kbase = ((size_t)(b * S_LEN + kb + sub) * NH + h) * DK;
        v8f c = zero_v8f();
#pragma unroll
        for (int c2 = 0; c2 < 2; ++c2) {
            v8bf lo = *(const v8bf*)&Kh[kbase + c2 * 32 + half * 8];
            v8bf hi = *(const v8bf*)&Kh[kbase + c2 * 32 + 16 + half * 8];
            c = wmma_bf16(qf[c2], combine16(lo, hi), c);
        }
#pragma unroll
        for (int r = 0; r < 8; ++r) {
            int m = half * 8 + r;
            scores[m * SMAX + kt * 16 + sub] = c[r] * 0.125f;  // 1/sqrt(64)
        }
    }
    __syncthreads();

    // Phase 2: per-row softmax with band mask (threads 0..15 own rows).
    if (tid < 16) {
        const int qi  = q0 + tid;
        const int jlo = max(0, qi - WIN) - kstart;
        const int jhi = min(S_LEN - 1, qi + WIN) - kstart;  // inclusive
        float mmax = -3.0e38f;
        for (int j = jlo; j <= jhi; ++j)
            mmax = fmaxf(mmax, scores[tid * SMAX + j]);
        float ssum = 0.0f;
        for (int j = jlo; j <= jhi; ++j) {
            float p = __expf(scores[tid * SMAX + j] - mmax);
            scores[tid * SMAX + j] = p;
            ssum += p;
        }
        const float inv = 1.0f / ssum;
        for (int j = 0; j < nk; ++j) {
            float p = (j >= jlo && j <= jhi) ? scores[tid * SMAX + j] * inv : 0.0f;
            scores[tid * SMAX + j] = p;
            pbuf[tid * SMAX + j]   = (__bf16)p;
        }
    }
    __syncthreads();

    // Phase 3: stream probabilities to the attn output (coalesced over j).
    {
        float* arow = attn + (((size_t)(b * NH + h) * S_LEN) + q0) * S_LEN;
        for (int idx = tid; idx < 16 * nk; idx += 128) {
            int m = idx / nk, j = idx - m * nk;
            arow[(size_t)m * S_LEN + kstart + j] = scores[m * SMAX + j];
        }
    }

    // Phase 4: context = P @ V. Each wave owns 16 d-columns.
    v8f cacc = zero_v8f();
    const int nchunks = nkp >> 5;
    for (int ch = 0; ch < nchunks; ++ch) {
        __syncthreads();
        for (int i = tid; i < 32 * 64; i += 128) {   // stage V chunk transposed
            int j = i >> 6, d = i & 63;
            int key = kstart + ch * 32 + j;
            __bf16 vv = (__bf16)0.0f;
            if (key < kend)
                vv = Vh[((size_t)(b * S_LEN + key) * NH + h) * DK + d];
            vts[d * 32 + j] = vv;
        }
        __syncthreads();
        v8bf plo = *(const v8bf*)&pbuf[sub * SMAX + ch * 32 + half * 8];
        v8bf phi = *(const v8bf*)&pbuf[sub * SMAX + ch * 32 + 16 + half * 8];
        int d = wave * 16 + sub;
        v8bf blo = *(const v8bf*)&vts[d * 32 + half * 8];
        v8bf bhi = *(const v8bf*)&vts[d * 32 + 16 + half * 8];
        cacc = wmma_bf16(combine16(plo, phi), combine16(blo, bhi), cacc);
    }

#pragma unroll
    for (int r = 0; r < 8; ++r) {
        int m = half * 8 + r;
        ctx[((size_t)(b * S_LEN + q0 + m) * NH + h) * DK + wave * 16 + sub] =
            (__bf16)cacc[r];
    }
}

// ---------------------------------------------------------------------------
extern "C" void kernel_launch(void* const* d_in, const int* in_sizes, int n_in,
                              void* d_out, int out_size, void* d_ws, size_t ws_size,
                              hipStream_t stream)
{
    (void)in_sizes; (void)n_in; (void)out_size; (void)ws_size;

    const float* q  = (const float*)d_in[0];
    const float* k  = (const float*)d_in[1];
    const float* v  = (const float*)d_in[2];
    const float* Wq = (const float*)d_in[3];
    const float* bq = (const float*)d_in[4];
    const float* Wk = (const float*)d_in[5];
    const float* bk = (const float*)d_in[6];
    const float* Wv = (const float*)d_in[7];
    const float* bv = (const float*)d_in[8];
    const float* Wo = (const float*)d_in[9];
    const float* bo = (const float*)d_in[10];

    const int B = 2, M = B * S_LEN, N = DMODEL, K = DMODEL;

    float* out  = (float*)d_out;                       // [B,S,D]
    float* attn = out + (size_t)M * N;                 // [B,H,S,S]

    // Workspace: head tensors + transposed bf16 weights (~40 MB).
    __bf16* Qh  = (__bf16*)d_ws;
    __bf16* Kh  = Qh + (size_t)M * N;
    __bf16* Vh  = Kh + (size_t)M * N;
    __bf16* Ctx = Vh + (size_t)M * N;
    __bf16* Wqt = Ctx + (size_t)M * N;
    __bf16* Wkt = Wqt + (size_t)N * K;
    __bf16* Wvt = Wkt + (size_t)N * K;
    __bf16* Wot = Wvt + (size_t)N * K;

    // bf16 copies of q/k/v live in the attn region of d_out: it is scratch
    // until the memset below (which runs after the QKV projections).
    __bf16* qbf = (__bf16*)attn;
    __bf16* kbf = qbf + (size_t)M * K;
    __bf16* vbf = kbf + (size_t)M * K;

    const int nElem = M * K;
    const int cvtBlocks = nElem / (256 * 4);
    cvt_bf16_kernel<<<cvtBlocks, 256, 0, stream>>>(q, qbf, nElem);
    cvt_bf16_kernel<<<cvtBlocks, 256, 0, stream>>>(k, kbf, nElem);
    cvt_bf16_kernel<<<cvtBlocks, 256, 0, stream>>>(v, vbf, nElem);

    dim3 gt(N / 32, K / 32);
    wt_kernel<<<gt, 256, 0, stream>>>(Wq, Wqt, N, K);
    wt_kernel<<<gt, 256, 0, stream>>>(Wk, Wkt, N, K);
    wt_kernel<<<gt, 256, 0, stream>>>(Wv, Wvt, N, K);
    wt_kernel<<<gt, 256, 0, stream>>>(Wo, Wot, N, K);

    dim3 gg(N / 128, M / 128);   // (8, 32)
    gemm_tdm_kernel<__bf16><<<gg, 256, 0, stream>>>(qbf, Wqt, bq, Qh, M, N, K);
    gemm_tdm_kernel<__bf16><<<gg, 256, 0, stream>>>(kbf, Wkt, bk, Kh, M, N, K);
    gemm_tdm_kernel<__bf16><<<gg, 256, 0, stream>>>(vbf, Wvt, bv, Vh, M, N, K);

    // attn tensor: zero everything once; the attention kernel writes the band.
    (void)hipMemsetAsync(attn, 0,
                         (size_t)B * NH * S_LEN * S_LEN * sizeof(float), stream);

    attn_kernel<<<B * NH * (S_LEN / 16), 128, 0, stream>>>(Qh, Kh, Vh, attn, Ctx);

    gemm_tdm_kernel<float><<<gg, 256, 0, stream>>>(Ctx, Wot, bo, out, M, N, K);
}
